// TransAttention_62483184222907
// MI455X (gfx1250) — compile-verified, matrix operand order checked by reading, not run on hardware
//
#include <hip/hip_runtime.h>
#include <hip/hip_bf16.h>

typedef __attribute__((ext_vector_type(16))) __bf16 v16bf;
typedef __attribute__((ext_vector_type(8)))  __bf16 v8bf;
typedef __attribute__((ext_vector_type(8)))  float  v8f;

#define B_ 4
#define L_ 1024
#define E_ 768
#define K_ 12
#define I_ 3072

__device__ __forceinline__ __bf16 f2bf(float f) { return (__bf16)f; }

// Build a 16x32 bf16 WMMA fragment for lane (r, hi):
// element e <-> K = (e>>3)*16 + hi*8 + (e&7); the two 8-element runs are
// contiguous in memory, so two 16B loads suffice.
__device__ __forceinline__ v16bf load_frag2(const __bf16* lo, const __bf16* hip_) {
  v8bf a = *(const v8bf*)lo;
  v8bf b = *(const v8bf*)hip_;
  v16bf f;
#pragma unroll
  for (int e = 0; e < 8; ++e) { f[e] = a[e]; f[e + 8] = b[e]; }
  return f;
}

__device__ __forceinline__ v8f wmma_bf16(v16bf a, v16bf b, v8f c) {
  // 8 args: (neg_a, A, neg_b, B, c_mod, C, reuse_a, reuse_b)
  return __builtin_amdgcn_wmma_f32_16x16x32_bf16(false, a, false, b, (short)0, c,
                                                 false, false);
}

// ---------------- prep kernels ----------------

__global__ void cvt_bf16_kernel(const float* __restrict__ s, __bf16* __restrict__ d, int n) {
  int i = blockIdx.x * blockDim.x + threadIdx.x;
  int st = gridDim.x * blockDim.x;
  for (; i < n; i += st) d[i] = f2bf(s[i]);
}

__global__ __launch_bounds__(256) void k2e_kernel(const float* __restrict__ x,
                                                  float* __restrict__ k2e) {
  __shared__ float red[256];
  int row = blockIdx.x;
  const float* xr = x + (size_t)row * E_;
  float s = 0.f;
  for (int e = threadIdx.x; e < E_; e += 256) { float v = xr[e]; s += v * v; }
  red[threadIdx.x] = s; __syncthreads();
  for (int st = 128; st > 0; st >>= 1) {
    if ((int)threadIdx.x < st) red[threadIdx.x] += red[threadIdx.x + st];
    __syncthreads();
  }
  if (threadIdx.x == 0) k2e[row] = red[0] * (1.0f / (float)E_);
}

// bias[b,k,m] = (2/E)*mu_k.x_m - |x_m|^2/E  (everything softmax needs besides XX)
__global__ __launch_bounds__(256) void muk2_kernel(const float* __restrict__ x,
                                                   const float* __restrict__ mu,
                                                   const float* __restrict__ k2e,
                                                   float* __restrict__ muko) {
  __shared__ float red[256];
  int idx = blockIdx.x;                 // b*K_*L_ + k*L_ + m
  int m = idx & (L_ - 1);
  int k = (idx >> 10) % K_;
  int b = idx / (K_ * L_);
  const float* xr = x + ((size_t)b * L_ + m) * E_;
  const float* mr = mu + (size_t)k * E_;
  float s = 0.f;
  for (int e = threadIdx.x; e < E_; e += 256) s += xr[e] * mr[e];
  red[threadIdx.x] = s; __syncthreads();
  for (int st = 128; st > 0; st >>= 1) {
    if ((int)threadIdx.x < st) red[threadIdx.x] += red[threadIdx.x + st];
    __syncthreads();
  }
  if (threadIdx.x == 0)
    muko[idx] = red[0] * (2.0f / (float)E_) - k2e[b * L_ + m];
}

// ---------------- bf16 WMMA GEMM (templated epilogue) ----------------
// C[M,N] = A[M,Kd] (bf16, row-major) @ B[Kd,N] (bf16, row-major) + bias
// EPI 1: bias + exact GELU (bf16 out)
// EPI 2: bias + residual add (f32 out)
// EPI 3: bias, bf16 out written TRANSPOSED per batch: out[(b*N+col)*L_ + m]
template <int EPI>
__global__ __launch_bounds__(256) void gemm_bf16_kernel(
    const __bf16* __restrict__ A, const __bf16* __restrict__ Bm,
    const float* __restrict__ bias, const float* __restrict__ res,
    float* __restrict__ outF, __bf16* __restrict__ outB, int N, int Kd) {
  __shared__ __bf16 As[128][32];
  __shared__ __bf16 BsT[128][40];  // [n][k], row = 80B = 5*16B (aligned)
  const int tid = threadIdx.x;
  const int wave = tid >> 5, lane = tid & 31;
  const int r = lane & 15, hi = lane >> 4;
  const int m0 = blockIdx.x * 128, n0 = blockIdx.y * 128;
  const int wm = (wave & 3) * 32, wn = (wave >> 2) * 64;

  const int rowA = tid >> 1, segA = (tid & 1) * 16;      // A staging coords
  const int krow = tid >> 3, cg = (tid & 7) * 16;        // B staging coords

  const v8f z8 = {0.f, 0.f, 0.f, 0.f, 0.f, 0.f, 0.f, 0.f};
  v8f acc[2][4];
#pragma unroll
  for (int i = 0; i < 2; ++i)
#pragma unroll
    for (int j = 0; j < 4; ++j) acc[i][j] = z8;

  for (int k0 = 0; k0 < Kd; k0 += 32) {
    __syncthreads();
    {  // stage A tile 128x32 (each thread: 16 contiguous bf16 = 2x uint4)
      const uint4* s = (const uint4*)(A + (size_t)(m0 + rowA) * Kd + k0 + segA);
      uint4* d = (uint4*)&As[rowA][segA];
      d[0] = s[0]; d[1] = s[1];
    }
    {  // stage B tile 32x128 transposed: read 16 along N, scatter into BsT[n][k]
      const __bf16* src = Bm + (size_t)(k0 + krow) * N + n0 + cg;
      v8bf q0 = *(const v8bf*)src;
      v8bf q1 = *(const v8bf*)(src + 8);
#pragma unroll
      for (int i = 0; i < 8; ++i) {
        BsT[cg + i][krow] = q0[i];
        BsT[cg + 8 + i][krow] = q1[i];
      }
    }
    if (k0 + 32 < Kd) {  // CDNA5 global_prefetch of next k-tile
      __builtin_prefetch(A + (size_t)(m0 + rowA) * Kd + k0 + 32 + segA, 0, 1);
      __builtin_prefetch(Bm + (size_t)(k0 + 32 + krow) * N + n0 + cg, 0, 1);
    }
    __syncthreads();
    v16bf af[2], bfc[4];
#pragma unroll
    for (int mt = 0; mt < 2; ++mt) {
      int row = wm + mt * 16 + r;
      af[mt] = load_frag2(&As[row][hi * 8], &As[row][16 + hi * 8]);
    }
#pragma unroll
    for (int nt = 0; nt < 4; ++nt) {
      int coln = wn + nt * 16 + r;
      bfc[nt] = load_frag2(&BsT[coln][hi * 8], &BsT[coln][16 + hi * 8]);
    }
#pragma unroll
    for (int mt = 0; mt < 2; ++mt)
#pragma unroll
      for (int nt = 0; nt < 4; ++nt)
        acc[mt][nt] = wmma_bf16(af[mt], bfc[nt], acc[mt][nt]);
  }

#pragma unroll
  for (int mt = 0; mt < 2; ++mt)
#pragma unroll
    for (int nt = 0; nt < 4; ++nt) {
      int col = n0 + wn + nt * 16 + r;
      float bv = bias[col];
#pragma unroll
      for (int j = 0; j < 8; ++j) {
        int row = m0 + wm + mt * 16 + hi * 8 + j;
        float v = acc[mt][nt][j] + bv;
        if (EPI == 1) v = 0.5f * v * (1.0f + erff(v * 0.70710678118f));
        if (EPI == 2) {
          size_t o = (size_t)row * N + col;
          outF[o] = v + res[o];
        }
        if (EPI == 1) outB[(size_t)row * N + col] = f2bf(v);
        if (EPI == 3) {  // transposed per-batch layout [b][col][m]
          size_t o = ((size_t)(row >> 10) * N + col) * L_ + (row & (L_ - 1));
          outB[o] = f2bf(v);
        }
      }
    }
}

// ---------------- fused distance-attention (flash-style) ----------------
// block = 256 threads (8 waves) = one b, one 16-query tile.
// phase A: waves jointly build XX[16 x 128] Gram tile via WMMA -> LDS
// phase B: heads distributed over waves (wave w: heads w, w+8); online
//          softmax per (row, head) across m, O += P@V via WMMA. V comes from
//          the transposed xvT[b][d][m] layout -> contiguous 16B frag loads.
__global__ __launch_bounds__(256) void attn_kernel(
    const float* __restrict__ x, const __bf16* __restrict__ x_bf,
    const __bf16* __restrict__ xvT, const float* __restrict__ mukc,
    float* __restrict__ hF, __bf16* __restrict__ hB) {
  __shared__ __bf16 Xq[16][E_];    // query rows, bf16 (24 KB)
  __shared__ float XXs[16][128];   // Gram tile handoff (8 KB)

  const int tid = threadIdx.x;
  const int wave = tid >> 5, lane = tid & 31;
  const int r = lane & 15, hi = lane >> 4;
  const int b = blockIdx.x >> 6;
  const int l0 = (blockIdx.x & 63) << 4;

  {  // stage 16 query rows (contiguous 16*768 bf16)
    const uint4* s = (const uint4*)(x_bf + (size_t)(b * L_ + l0) * E_);
    uint4* d = (uint4*)&Xq[0][0];
    for (int i = tid; i < 16 * E_ / 8; i += 256) d[i] = s[i];
  }

  const v8f z8 = {0.f, 0.f, 0.f, 0.f, 0.f, 0.f, 0.f, 0.f};
  v8f O[2][4];
#pragma unroll
  for (int a = 0; a < 2; ++a)
#pragma unroll
    for (int c = 0; c < 4; ++c) O[a][c] = z8;
  float Mrow[2] = {-1e30f, -1e30f};
  float Srow[2] = {0.f, 0.f};
  const float c2 = 2.0f / (float)E_;
  const __bf16* xr = x_bf + (size_t)(b * L_) * E_;

  for (int m0 = 0; m0 < L_; m0 += 128) {
    __syncthreads();  // XXs free (prev phase B done) + Xq staged (1st iter)
    // ---- phase A: XX columns [m0 + wave*16 .. +15]
    const __bf16* brow = xr + (size_t)(m0 + wave * 16 + r) * E_;
    if (m0 + 128 < L_)
      __builtin_prefetch(xr + (size_t)(m0 + 128 + wave * 16 + r) * E_, 0, 1);
    v8f acc = z8;
    for (int kk = 0; kk < E_; kk += 32) {
      v16bf a = load_frag2(&Xq[r][kk + hi * 8], &Xq[r][kk + 16 + hi * 8]);
      v16bf bb = load_frag2(brow + kk + hi * 8, brow + kk + 16 + hi * 8);
      acc = wmma_bf16(a, bb, acc);
    }
#pragma unroll
    for (int j = 0; j < 8; ++j) XXs[hi * 8 + j][wave * 16 + r] = acc[j];
    __syncthreads();
    // ---- phase B: per-head online softmax + P@V
    for (int h = wave, hs = 0; h < K_; h += 8, ++hs) {
      const float* mukp = mukc + ((size_t)b * K_ + h) * L_;
      for (int mc = 0; mc < 128; mc += 32) {
        // scores: contiguous 8-float runs matching the fragment K-layout
        v8f xxlo = *(const v8f*)&XXs[r][mc + hi * 8];
        v8f xxhi = *(const v8f*)&XXs[r][mc + 16 + hi * 8];
        v8f mklo = *(const v8f*)(mukp + m0 + mc + hi * 8);
        v8f mkhi = *(const v8f*)(mukp + m0 + mc + 16 + hi * 8);
        float sv[16];
        float mymax = -1e30f;
#pragma unroll
        for (int e = 0; e < 8; ++e) {
          sv[e] = c2 * xxlo[e] + mklo[e];
          sv[e + 8] = c2 * xxhi[e] + mkhi[e];
          mymax = fmaxf(mymax, fmaxf(sv[e], sv[e + 8]));
        }
        mymax = fmaxf(mymax, __shfl_xor(mymax, 16, 32));  // wave32 half swap
        float newM = fmaxf(Mrow[hs], mymax);
        float ps = 0.f;
        v16bf pf;
#pragma unroll
        for (int e = 0; e < 16; ++e) {
          float p = __expf(sv[e] - newM);
          ps += p;
          pf[e] = f2bf(p);
        }
        ps += __shfl_xor(ps, 16, 32);
        float fac = __expf(Mrow[hs] - newM);
        Srow[hs] = Srow[hs] * fac + ps;
        Mrow[hs] = newM;
        float fj[8];
#pragma unroll
        for (int j = 0; j < 8; ++j) fj[j] = __shfl(fac, hi * 8 + j, 32);
#pragma unroll
        for (int nb = 0; nb < 4; ++nb) {
#pragma unroll
          for (int j = 0; j < 8; ++j) O[hs][nb][j] *= fj[j];
          // V frag: xvT[b][d][m], d = h*64 + nb*16 + r, m runs contiguous
          const __bf16* vcol =
              xvT + ((size_t)b * E_ + h * 64 + nb * 16 + r) * L_ + m0 + mc;
          v16bf vf = load_frag2(vcol + hi * 8, vcol + 16 + hi * 8);
          O[hs][nb] = wmma_bf16(pf, vf, O[hs][nb]);
        }
      }
    }
  }
  // ---- finalize: h = x + O/S
  for (int h = wave, hs = 0; h < K_; h += 8, ++hs) {
    float sj[8];
#pragma unroll
    for (int j = 0; j < 8; ++j) sj[j] = __shfl(Srow[hs], hi * 8 + j, 32);
#pragma unroll
    for (int nb = 0; nb < 4; ++nb)
#pragma unroll
      for (int j = 0; j < 8; ++j) {
        int row = l0 + hi * 8 + j;
        int col = h * 64 + nb * 16 + r;
        size_t o = ((size_t)b * L_ + row) * E_ + col;
        float hv = x[o] + O[hs][nb][j] / sj[j];
        hF[o] = hv;
        hB[o] = f2bf(hv);
      }
  }
}

// ---------------- LayerNorm ----------------
__global__ __launch_bounds__(256) void ln_kernel(const float* __restrict__ y,
                                                 const float* __restrict__ g,
                                                 const float* __restrict__ bt,
                                                 float* __restrict__ out) {
  __shared__ float red[256], red2[256];
  int row = blockIdx.x;
  const float* yr = y + (size_t)row * E_;
  float s = 0.f, s2 = 0.f;
  for (int e = threadIdx.x; e < E_; e += 256) {
    float v = yr[e];
    s += v; s2 += v * v;
  }
  red[threadIdx.x] = s; red2[threadIdx.x] = s2;
  __syncthreads();
  for (int st = 128; st > 0; st >>= 1) {
    if ((int)threadIdx.x < st) {
      red[threadIdx.x] += red[threadIdx.x + st];
      red2[threadIdx.x] += red2[threadIdx.x + st];
    }
    __syncthreads();
  }
  float mean = red[0] * (1.0f / (float)E_);
  float var = red2[0] * (1.0f / (float)E_) - mean * mean;
  float rs = rsqrtf(var + 1e-12f);
  for (int e = threadIdx.x; e < E_; e += 256)
    out[(size_t)row * E_ + e] = (yr[e] - mean) * rs * g[e] + bt[e];
}

// ---------------- launch ----------------
extern "C" void kernel_launch(void* const* d_in, const int* in_sizes, int n_in,
                              void* d_out, int out_size, void* d_ws,
                              size_t ws_size, hipStream_t stream) {
  const float* x    = (const float*)d_in[0];
  const float* mu   = (const float*)d_in[1];
  const float* valw = (const float*)d_in[2];
  const float* valb = (const float*)d_in[3];
  const float* w1   = (const float*)d_in[4];
  const float* b1   = (const float*)d_in[5];
  const float* w2   = (const float*)d_in[6];
  const float* b2   = (const float*)d_in[7];
  const float* lng  = (const float*)d_in[8];
  const float* lnb  = (const float*)d_in[9];
  float* out = (float*)d_out;

  const size_t ML = (size_t)B_ * L_;  // 4096 rows
  char* w = (char*)d_ws;
  size_t off = 0;
  auto take = [&](size_t bytes) -> char* {
    char* p = w + off;
    off = (off + bytes + 255) & ~(size_t)255;
    return p;
  };
  __bf16* x_bf  = (__bf16*)take(ML * E_ * 2);
  __bf16* vw_bf = (__bf16*)take((size_t)E_ * E_ * 2);
  __bf16* w1_bf = (__bf16*)take((size_t)E_ * I_ * 2);
  __bf16* w2_bf = (__bf16*)take((size_t)I_ * E_ * 2);
  __bf16* xvT   = (__bf16*)take(ML * E_ * 2);   // transposed [b][d][m]
  __bf16* h_bf  = (__bf16*)take(ML * E_ * 2);
  float*  h_f   = (float*)take(ML * E_ * 4);
  __bf16* in_bf = (__bf16*)take(ML * I_ * 2);
  float*  y_f   = (float*)take(ML * E_ * 4);
  float*  k2e   = (float*)take(ML * 4);
  float*  mukc  = (float*)take((size_t)B_ * K_ * L_ * 4);

  cvt_bf16_kernel<<<4096, 256, 0, stream>>>(x, x_bf, (int)(ML * E_));
  cvt_bf16_kernel<<<2048, 256, 0, stream>>>(valw, vw_bf, E_ * E_);
  cvt_bf16_kernel<<<4096, 256, 0, stream>>>(w1, w1_bf, E_ * I_);
  cvt_bf16_kernel<<<4096, 256, 0, stream>>>(w2, w2_bf, I_ * E_);
  k2e_kernel<<<(int)ML, 256, 0, stream>>>(x, k2e);
  muk2_kernel<<<B_ * K_ * L_, 256, 0, stream>>>(x, mu, k2e, mukc);

  // xv = x @ val_w + val_b  -> bf16, transposed [b][d][m]
  gemm_bf16_kernel<3><<<dim3(ML / 128, E_ / 128), 256, 0, stream>>>(
      x_bf, vw_bf, valb, nullptr, nullptr, xvT, E_, E_);
  // attention -> h (f32 + bf16)
  attn_kernel<<<B_ * (L_ / 16), 256, 0, stream>>>(x, x_bf, xvT, mukc, h_f,
                                                  h_bf);
  // inter = gelu(h @ w1 + b1)  (bf16 out)
  gemm_bf16_kernel<1><<<dim3(ML / 128, I_ / 128), 256, 0, stream>>>(
      h_bf, w1_bf, b1, nullptr, nullptr, in_bf, I_, E_);
  // y = inter @ w2 + b2 + h   (f32 out)
  gemm_bf16_kernel<2><<<dim3(ML / 128, E_ / 128), 256, 0, stream>>>(
      in_bf, w2_bf, b2, h_f, y_f, nullptr, E_, I_);
  // layernorm
  ln_kernel<<<(int)ML, 256, 0, stream>>>(y_f, lng, lnb, out);

  (void)in_sizes; (void)n_in; (void)out_size; (void)ws_size;
}
